// AttentionLSTM_85813446574180
// MI455X (gfx1250) — matrix operand order repaired and checked by reading, not compiled, not run on hardware
//
#include <hip/hip_runtime.h>
#include <hip/hip_bf16.h>
#include <cstdint>

typedef _Float16 f16;
typedef __attribute__((ext_vector_type(16))) _Float16 v16h;
typedef __attribute__((ext_vector_type(8)))  float    v8f;

#define DEV __device__ __forceinline__

DEV float sigf(float x) { return 1.0f / (1.0f + __expf(-x)); }

// ---------------- WMMA tile helpers (CDNA5 wave32 layouts, CDNA5 ISA) ----------------
// A 16x32 f16 (MxK), row-major source, lda in elements.
// lane<16: M=lane, K in {0..7}u{16..23}; lane>=16: M=lane-16, K in {8..15}u{24..31}.
DEV v16h load_a16(const f16* base, int lda) {
  const int lane = threadIdx.x & 31;
  const int hf   = lane >> 4;
  const int m    = lane & 15;
  const uint32_t* row = reinterpret_cast<const uint32_t*>(base + (size_t)m * lda);
  v16h a;
#pragma unroll
  for (int i = 0; i < 16; i += 2) {
    const int k = (i & 7) + ((i >> 3) << 4) + (hf << 3);
    union { uint32_t u; f16 h[2]; } cv;
    cv.u = row[k >> 1];
    a[i] = cv.h[0]; a[i + 1] = cv.h[1];
  }
  return a;
}
// B 32x16 f16 (KxN); source is weight row-major W[n][k], ldb in elements.
// lane<16: N=lane, K in {0..7}u{16..23}; lane>=16: N=lane-16, K in {8..15}u{24..31}.
DEV v16h load_b16(const f16* base, int ldb) {
  const int lane = threadIdx.x & 31;
  const int hf   = lane >> 4;
  const int n    = lane & 15;
  const uint32_t* row = reinterpret_cast<const uint32_t*>(base + (size_t)n * ldb + (hf << 3));
  v16h b;
#pragma unroll
  for (int i = 0; i < 16; i += 2) {
    union { uint32_t u; f16 h[2]; } cv;
    cv.u = row[((i & 7) + ((i >> 3) << 4)) >> 1];
    b[i] = cv.h[0]; b[i + 1] = cv.h[1];
  }
  return b;
}
// C/D f32 16x16: VGPR r holds M=r+8*(lane>>4), N=lane&15.
DEV void store_d_f32(float* base, int ldd, v8f c) {
  const int lane = threadIdx.x & 31;
  const int hf   = lane >> 4;
  const int n    = lane & 15;
#pragma unroll
  for (int r = 0; r < 8; ++r) base[(size_t)(r + hf * 8) * ldd + n] = c[r];
}
DEV void store_d_f16(f16* base, int ldd, v8f c) {
  const int lane = threadIdx.x & 31;
  const int hf   = lane >> 4;
  const int n    = lane & 15;
#pragma unroll
  for (int r = 0; r < 8; ++r) base[(size_t)(r + hf * 8) * ldd + n] = (f16)c[r];
}
DEV v8f wmma16(v16h a, v16h b, v8f c) {
  return __builtin_amdgcn_wmma_f32_16x16x32_f16(false, a, false, b, (short)0, c, false, false);
}

// ---------------- prep kernels ----------------
__global__ void conv_f16(const float* __restrict__ src, f16* __restrict__ dst, int n) {
  int i = blockIdx.x * 256 + threadIdx.x;
  if (i < n) dst[i] = (f16)src[i];
}

// decoder combined weight: [dec_Wih(1024x536) | dec_Whh(1024x256) | pad 8] -> 1024x800 f16
__global__ void prep_decw(const float* __restrict__ Wih, const float* __restrict__ Whh,
                          f16* __restrict__ dst) {
  int i = blockIdx.x * 256 + threadIdx.x;
  if (i >= 1024 * 800) return;
  int n = i / 800, k = i % 800;
  float v = 0.0f;
  if (k < 536)      v = Wih[(size_t)n * 536 + k];
  else if (k < 792) v = Whh[(size_t)n * 256 + (k - 536)];
  dst[i] = (f16)v;
}

// lang_active[b,l] = (sum_{s,d} |x|) > 0
__global__ void prep_act(const float* __restrict__ x, float* __restrict__ act) {
  __shared__ float red[256];
  const int m = blockIdx.x, b = m / 5, l = m % 5;
  float a = 0.0f;
  for (int i = threadIdx.x; i < 128 * 24; i += 256) {
    int s = i / 24, d = i % 24;
    a += fabsf(x[(((size_t)b * 128 + s) * 5 + l) * 24 + d]);
  }
  red[threadIdx.x] = a; __syncthreads();
  for (int s2 = 128; s2 > 0; s2 >>= 1) {
    if (threadIdx.x < s2) red[threadIdx.x] += red[threadIdx.x + s2];
    __syncthreads();
  }
  if (threadIdx.x == 0) act[m] = (red[0] > 0.0f) ? 1.0f : 0.0f;
}

// Xf16[s*320+m][0..31] = [ x[b,s,l,:24] | lang_embed[l]*active[b,l] ]
__global__ void prep_x(const float* __restrict__ x, const float* __restrict__ emb,
                       const float* __restrict__ act, f16* __restrict__ X) {
  int i = blockIdx.x * 256 + threadIdx.x;          // 128*320*32 exact
  int k = i & 31;
  int r = i >> 5;
  int m = r % 320, s = r / 320;
  int b = m / 5, l = m % 5;
  float v;
  if (k < 24) v = x[(((size_t)b * 128 + s) * 5 + l) * 24 + k];
  else        v = emb[l * 8 + (k - 24)] * act[m];
  X[i] = (f16)v;
}

__global__ void init_state(float* h, float* c, f16* hF) {
  int i = blockIdx.x * 256 + threadIdx.x;          // 2*320*256 exact
  h[i] = 0.0f; c[i] = 0.0f; hF[i] = (f16)0.0f;
}

// ---------------- encoder recurrent step (1x4 register-blocked WMMA) ----------------
// gates[dir][320][1024] = bias + [x_t | h] @ [Wih | Whh]^T  (K = 32 + 256)
// wave computes a 16x64 strip: A fragment loaded once per k-step, reused over 4 B tiles.
__global__ void enc_step_gemm(const f16* __restrict__ X, const f16* __restrict__ Wih,
                              const f16* __restrict__ Whh, const float* __restrict__ bF,
                              const float* __restrict__ bB, const f16* __restrict__ hF,
                              float* __restrict__ gates, int step) {
  const int dir  = blockIdx.z;
  const int t    = dir ? (127 - step) : step;
  const int wave = threadIdx.x >> 5;
  const int job  = blockIdx.x * 4 + wave;          // 0..319 (20 Mt x 16 Ngroups)
  const int mt = job >> 4, ng = job & 15;
  const int m0 = mt * 16, n0 = ng * 64;
  const int nn = threadIdx.x & 15;
  const float* bias = dir ? bB : bF;
  v8f acc[4];
#pragma unroll
  for (int j = 0; j < 4; ++j) {
    const float bv = bias[n0 + j * 16 + nn];
#pragma unroll
    for (int r = 0; r < 8; ++r) acc[j][r] = bv;
  }
  {  // K-step 0: x_t part (K=32)
    v16h a = load_a16(X + ((size_t)t * 320 + m0) * 32, 32);
    const f16* wb = Wih + (size_t)dir * 1024 * 32 + (size_t)n0 * 32;
#pragma unroll
    for (int j = 0; j < 4; ++j) {
      v16h b = load_b16(wb + (size_t)j * 16 * 32, 32);
      acc[j] = wmma16(a, b, acc[j]);
    }
  }
  const f16* hb = hF  + (size_t)dir * 320 * 256  + (size_t)m0 * 256;
  const f16* wb = Whh + (size_t)dir * 1024 * 256 + (size_t)n0 * 256;
#pragma unroll
  for (int ks = 0; ks < 8; ++ks) {   // recurrent part (K=256)
    v16h a = load_a16(hb + ks * 32, 256);
#pragma unroll
    for (int j = 0; j < 4; ++j) {
      v16h b = load_b16(wb + (size_t)j * 16 * 256 + ks * 32, 256);
      acc[j] = wmma16(a, b, acc[j]);
    }
  }
  float* dst = gates + (size_t)dir * 320 * 1024 + (size_t)m0 * 1024 + n0;
#pragma unroll
  for (int j = 0; j < 4; ++j) store_d_f32(dst + j * 16, 1024, acc[j]);
}

__global__ void enc_step_point(const float* __restrict__ gates, float* __restrict__ h,
                               float* __restrict__ c, f16* __restrict__ hF,
                               f16* __restrict__ encOut, int step) {
  int idx = blockIdx.x * 256 + threadIdx.x;        // 2*320*256 exact
  int dir = idx / 81920;
  int rem = idx - dir * 81920;
  int m = rem >> 8, n = rem & 255;
  const float* g = gates + (size_t)dir * 320 * 1024 + (size_t)m * 1024;
  float gi = g[n], gf = g[n + 256], gg = g[n + 512], go = g[n + 768];
  size_t o = (size_t)dir * 81920 + (size_t)m * 256 + n;
  float cn = sigf(gf) * c[o] + sigf(gi) * tanhf(gg);
  float hn = sigf(go) * tanhf(cn);
  c[o] = cn; h[o] = hn; hF[o] = (f16)hn;
  int s = dir ? (127 - step) : step;
  int b = m / 5, l = m % 5;
  encOut[((size_t)b * 640 + l * 128 + s) * 512 + dir * 256 + n] = (f16)hn;
}

// ---------------- bridge ----------------
__global__ void bridge_cat(const float* __restrict__ h, const float* __restrict__ c,
                           float* __restrict__ hcat, float* __restrict__ ccat) {
  int idx = blockIdx.x * 256 + threadIdx.x;        // 64*512 exact
  int b = idx >> 9, k = idx & 511;
  int dir = k >> 8, kk = k & 255;
  float sh = 0.0f, sc2 = 0.0f;
  for (int l = 0; l < 5; ++l) {
    size_t o = (size_t)dir * 81920 + (size_t)(b * 5 + l) * 256 + kk;
    sh += h[o]; sc2 += c[o];
  }
  hcat[idx] = sh * 0.2f; ccat[idx] = sc2 * 0.2f;
}

__global__ void bridge_proj(const float* __restrict__ hcat, const float* __restrict__ ccat,
                            const float* __restrict__ hbW, const float* __restrict__ hbB,
                            const float* __restrict__ cbW, const float* __restrict__ cbB,
                            float* __restrict__ hdec, float* __restrict__ cdec,
                            float* __restrict__ curr) {
  int idx = blockIdx.x * 256 + threadIdx.x;        // 64*256 exact
  int b = idx >> 8, n = idx & 255;
  const float* hc = hcat + (size_t)b * 512;
  const float* cc = ccat + (size_t)b * 512;
  const float* wh = hbW + (size_t)n * 512;
  const float* wc = cbW + (size_t)n * 512;
  float ah = hbB[n], ac = cbB[n];
  for (int k = 0; k < 512; ++k) { ah += hc[k] * wh[k]; ac += cc[k] * wc[k]; }
  hdec[idx] = ah; cdec[idx] = ac;
  if (n < 24) curr[b * 24 + n] = 0.0f;
}

// ---------------- keys projection: kp = enc_out @ Wk^T + b (40960x512, K=512) ----------------
// 1x4 register blocking: wave computes 16x64 strip. 2560 Mt x 8 Ngroups = 20480 waves.
__global__ void keys_gemm(const f16* __restrict__ encOut, const f16* __restrict__ WkF,
                          const float* __restrict__ Wkb, f16* __restrict__ kp) {
  const int wave = threadIdx.x >> 5;
  const int job  = blockIdx.x * 4 + wave;
  const int mt = job >> 3, ng = job & 7;
  const int m0 = mt * 16, n0 = ng * 64;
  const int nn = threadIdx.x & 15;
  v8f acc[4];
#pragma unroll
  for (int j = 0; j < 4; ++j) {
    const float bv = Wkb[n0 + j * 16 + nn];
#pragma unroll
    for (int r = 0; r < 8; ++r) acc[j][r] = bv;
  }
  const f16* ab = encOut + (size_t)m0 * 512;
  const f16* bb = WkF    + (size_t)n0 * 512;
#pragma unroll 2
  for (int ks = 0; ks < 16; ++ks) {
    v16h a = load_a16(ab + ks * 32, 512);
#pragma unroll
    for (int j = 0; j < 4; ++j) {
      v16h b = load_b16(bb + (size_t)j * 16 * 512 + ks * 32, 512);
      acc[j] = wmma16(a, b, acc[j]);
    }
  }
  f16* dst = kp + (size_t)m0 * 512 + n0;
#pragma unroll
  for (int j = 0; j < 4; ++j) store_d_f16(dst + j * 16, 512, acc[j]);
}

// ---------------- decoder step ----------------
// one block per batch row: q, Bahdanau scores, softmax, context; builds inp800 f16
__global__ void dec_attn(const float* __restrict__ hdec, const float* __restrict__ curr,
                         const float* __restrict__ WqW, const float* __restrict__ Wqb,
                         const f16* __restrict__ kp, const f16* __restrict__ encOut,
                         const float* __restrict__ vW, const float* __restrict__ vb,
                         f16* __restrict__ inp) {
  __shared__ float q_s[512];
  __shared__ float sc[640];
  __shared__ float red[256];
  const int b = blockIdx.x, tid = threadIdx.x;
  const float* h = hdec + (size_t)b * 256;
  for (int d = tid; d < 512; d += 256) {
    const float* w = WqW + (size_t)d * 256;
    float a = Wqb[d];
    for (int k = 0; k < 256; ++k) a += h[k] * w[k];
    q_s[d] = a;
  }
  __syncthreads();
  for (int k = tid; k < 640; k += 256) {
    const f16* kr = kp + ((size_t)b * 640 + k) * 512;
    float a = vb[0];
    for (int d = 0; d < 512; ++d) a += vW[d] * tanhf(q_s[d] + (float)kr[d]);
    sc[k] = a;
  }
  __syncthreads();
  float lm = -3.0e38f;
  for (int k = tid; k < 640; k += 256) lm = fmaxf(lm, sc[k]);
  red[tid] = lm; __syncthreads();
  for (int s2 = 128; s2 > 0; s2 >>= 1) {
    if (tid < s2) red[tid] = fmaxf(red[tid], red[tid + s2]);
    __syncthreads();
  }
  float mx = red[0]; __syncthreads();
  float ls = 0.0f;
  for (int k = tid; k < 640; k += 256) { float e = __expf(sc[k] - mx); sc[k] = e; ls += e; }
  red[tid] = ls; __syncthreads();
  for (int s2 = 128; s2 > 0; s2 >>= 1) {
    if (tid < s2) red[tid] += red[tid + s2];
    __syncthreads();
  }
  float inv = 1.0f / red[0]; __syncthreads();
  for (int k = tid; k < 640; k += 256) sc[k] *= inv;
  __syncthreads();
  f16* row = inp + (size_t)b * 800;
  for (int d = tid; d < 512; d += 256) {
    const f16* eo = encOut + (size_t)b * 640 * 512 + d;
    float a = 0.0f;
    for (int k = 0; k < 640; ++k) a += sc[k] * (float)eo[(size_t)k * 512];
    row[24 + d] = (f16)a;
  }
  for (int i = tid; i < 24; i += 256) row[i] = (f16)curr[b * 24 + i];
  row[536 + tid] = (f16)h[tid];
  if (tid < 8) row[792 + tid] = (f16)0.0f;
}

// gatesD[64][1024] = bias + inp800 @ decW^T (K=800 -> 25 wmma k-steps)
// 1x4 register blocking: 4 Mt x 16 Ngroups = 64 waves = 16 blocks.
__global__ void dec_gemm(const f16* __restrict__ inp, const f16* __restrict__ decW,
                         const float* __restrict__ decB, float* __restrict__ gates) {
  const int wave = threadIdx.x >> 5;
  const int job  = blockIdx.x * 4 + wave;
  const int mt = job >> 4, ng = job & 15;
  const int m0 = mt * 16, n0 = ng * 64;
  const int nn = threadIdx.x & 15;
  v8f acc[4];
#pragma unroll
  for (int j = 0; j < 4; ++j) {
    const float bv = decB[n0 + j * 16 + nn];
#pragma unroll
    for (int r = 0; r < 8; ++r) acc[j][r] = bv;
  }
  const f16* ab = inp  + (size_t)m0 * 800;
  const f16* bb = decW + (size_t)n0 * 800;
#pragma unroll 5
  for (int ks = 0; ks < 25; ++ks) {
    v16h a = load_a16(ab + ks * 32, 800);
#pragma unroll
    for (int j = 0; j < 4; ++j) {
      v16h b = load_b16(bb + (size_t)j * 16 * 800 + ks * 32, 800);
      acc[j] = wmma16(a, b, acc[j]);
    }
  }
  float* dst = gates + (size_t)m0 * 1024 + n0;
#pragma unroll
  for (int j = 0; j < 4; ++j) store_d_f32(dst + j * 16, 1024, acc[j]);
}

// LSTM pointwise + fc head; one block per batch row (h staged in LDS for fc)
__global__ void dec_point_fc(const float* __restrict__ gates, float* __restrict__ hdec,
                             float* __restrict__ cdec, const float* __restrict__ fcW,
                             const float* __restrict__ fcB, float* __restrict__ curr,
                             float* __restrict__ out, int step) {
  __shared__ float hs[256];
  const int b = blockIdx.x, n = threadIdx.x;
  const float* g = gates + (size_t)b * 1024;
  float gi = g[n], gf = g[n + 256], gg = g[n + 512], go = g[n + 768];
  size_t o = (size_t)b * 256 + n;
  float cn = sigf(gf) * cdec[o] + sigf(gi) * tanhf(gg);
  float hn = sigf(go) * tanhf(cn);
  cdec[o] = cn; hdec[o] = hn; hs[n] = hn;
  __syncthreads();
  if (n < 25) {
    const float* w = fcW + (size_t)n * 256;
    float a = fcB[n];
    for (int k = 0; k < 256; ++k) a += hs[k] * w[k];
    float pv;
    if (n < 24) { pv = tanhf(a); curr[b * 24 + n] = pv; }
    else        { pv = sigf(a); }
    out[(size_t)b * 3200 + (size_t)step * 25 + n] = pv;
  }
}

// ---------------- launcher ----------------
extern "C" void kernel_launch(void* const* d_in, const int* in_sizes, int n_in,
                              void* d_out, int out_size, void* d_ws, size_t ws_size,
                              hipStream_t stream) {
  (void)in_sizes; (void)n_in; (void)out_size; (void)ws_size;
  const float* x    = (const float*)d_in[0];
  const float* lang = (const float*)d_in[1];
  const float* eWihF = (const float*)d_in[2];
  const float* eWhhF = (const float*)d_in[3];
  const float* ebF   = (const float*)d_in[4];
  const float* eWihB = (const float*)d_in[5];
  const float* eWhhB = (const float*)d_in[6];
  const float* ebB   = (const float*)d_in[7];
  const float* hbW = (const float*)d_in[8];
  const float* hbB = (const float*)d_in[9];
  const float* cbW = (const float*)d_in[10];
  const float* cbB = (const float*)d_in[11];
  const float* WqW = (const float*)d_in[12];
  const float* Wqb = (const float*)d_in[13];
  const float* WkW = (const float*)d_in[14];
  const float* Wkb = (const float*)d_in[15];
  const float* vW  = (const float*)d_in[16];
  const float* vb  = (const float*)d_in[17];
  const float* dWih = (const float*)d_in[18];
  const float* dWhh = (const float*)d_in[19];
  const float* db   = (const float*)d_in[20];
  const float* fcW  = (const float*)d_in[21];
  const float* fcB  = (const float*)d_in[22];
  float* out = (float*)d_out;

  char* w = (char*)d_ws;
  auto alloc = [&](size_t bytes) -> char* {
    char* p = w; w += (bytes + 255) & ~(size_t)255; return p;
  };
  f16*   Xf      = (f16*)  alloc(sizeof(f16) * 128 * 320 * 32);
  f16*   WihF16  = (f16*)  alloc(sizeof(f16) * 2 * 1024 * 32);
  f16*   WhhF16  = (f16*)  alloc(sizeof(f16) * 2 * 1024 * 256);
  f16*   WkF16   = (f16*)  alloc(sizeof(f16) * 512 * 512);
  f16*   decWF16 = (f16*)  alloc(sizeof(f16) * 1024 * 800);
  float* act     = (float*)alloc(sizeof(float) * 320);
  float* henc    = (float*)alloc(sizeof(float) * 2 * 320 * 256);
  float* cenc    = (float*)alloc(sizeof(float) * 2 * 320 * 256);
  f16*   hencF   = (f16*)  alloc(sizeof(f16) * 2 * 320 * 256);
  float* gatesE  = (float*)alloc(sizeof(float) * 2 * 320 * 1024);
  f16*   encOut  = (f16*)  alloc(sizeof(f16) * (size_t)64 * 640 * 512);
  f16*   kp      = (f16*)  alloc(sizeof(f16) * (size_t)64 * 640 * 512);
  float* hcat    = (float*)alloc(sizeof(float) * 64 * 512);
  float* ccat    = (float*)alloc(sizeof(float) * 64 * 512);
  float* hdec    = (float*)alloc(sizeof(float) * 64 * 256);
  float* cdec    = (float*)alloc(sizeof(float) * 64 * 256);
  float* curr    = (float*)alloc(sizeof(float) * 64 * 24);
  f16*   inp     = (f16*)  alloc(sizeof(f16) * 64 * 800);
  float* gatesD  = (float*)alloc(sizeof(float) * 64 * 1024);

  // one-time prep
  conv_f16<<<128,  256, 0, stream>>>(eWihF, WihF16,          32768);
  conv_f16<<<128,  256, 0, stream>>>(eWihB, WihF16 + 32768,  32768);
  conv_f16<<<1024, 256, 0, stream>>>(eWhhF, WhhF16,          262144);
  conv_f16<<<1024, 256, 0, stream>>>(eWhhB, WhhF16 + 262144, 262144);
  conv_f16<<<1024, 256, 0, stream>>>(WkW,   WkF16,           262144);
  prep_decw<<<3200, 256, 0, stream>>>(dWih, dWhh, decWF16);
  prep_act<<<320, 256, 0, stream>>>(x, act);
  prep_x<<<5120, 256, 0, stream>>>(x, lang, act, Xf);
  init_state<<<640, 256, 0, stream>>>(henc, cenc, hencF);

  // bidirectional encoder recurrence (fwd + bwd fused per step)
  for (int step = 0; step < 128; ++step) {
    enc_step_gemm<<<dim3(80, 1, 2), 128, 0, stream>>>(Xf, WihF16, WhhF16, ebF, ebB,
                                                      hencF, gatesE, step);
    enc_step_point<<<640, 256, 0, stream>>>(gatesE, henc, cenc, hencF, encOut, step);
  }

  bridge_cat<<<128, 256, 0, stream>>>(henc, cenc, hcat, ccat);
  bridge_proj<<<64, 256, 0, stream>>>(hcat, ccat, hbW, hbB, cbW, cbB, hdec, cdec, curr);
  keys_gemm<<<5120, 128, 0, stream>>>(encOut, WkF16, Wkb, kp);

  // attention decoder recurrence
  for (int step = 0; step < 128; ++step) {
    dec_attn<<<64, 256, 0, stream>>>(hdec, curr, WqW, Wqb, kp, encOut, vW, vb, inp);
    dec_gemm<<<16, 128, 0, stream>>>(inp, decWF16, db, gatesD);
    dec_point_fc<<<64, 256, 0, stream>>>(gatesD, hdec, cdec, fcW, fcB, curr, out, step);
  }
}
